// Attention_20633022890922
// MI455X (gfx1250) — matrix operand order repaired and checked by reading, not compiled, not run on hardware
//
#include <hip/hip_runtime.h>
#include <hip/hip_bf16.h>
#include <stdint.h>

// out = tokens @ W_eff + bo  (exact algebraic collapse of the reference:
// softmax rows sum to 1, so the QK/softmax path cancels).
//
// M=65536 (B*N), K=512, N=512.  f16 WMMA, f32 accumulate, double-buffered
// async global->LDS staging, K-loop unrolled x2 for compile-time buffers.

typedef __attribute__((ext_vector_type(16))) _Float16 v16h;
typedef __attribute__((ext_vector_type(8)))  float    v8f;

#define MTOT   65536
#define KDIM   512
#define NDIM   512
#define BM     64       // block tile M (2 waves)
#define BN     256      // block tile N (4 waves)
#define KSTEP  32       // WMMA K per step
#define LDSW   33       // padded LDS row stride (floats) -> conflict-free
#define BUFELT (BM * LDSW)

// ---------- prep 1: head-summed projection matrices -------------------------
__global__ void attn_prep_sums(const float* __restrict__ Wv,
                               const float* __restrict__ Wo,
                               float* __restrict__ WvSum,   // [512][64]
                               float* __restrict__ WoSum) { // [64][512]
  int t = blockIdx.x * blockDim.x + threadIdx.x;   // 65536 threads
  if (t < 512 * 64) {
    int h = t >> 6, d = t & 63;
    float s = 0.f;
    #pragma unroll
    for (int v = 0; v < 8; ++v) s += Wv[h * 512 + v * 64 + d];
    WvSum[t] = s;
  } else {
    int u = t - 512 * 64;
    int d = u >> 9, j = u & 511;
    float s = 0.f;
    #pragma unroll
    for (int q = 0; q < 8; ++q) s += Wo[(q * 64 + d) * 512 + j];
    WoSum[u] = s;
  }
}

// ---------- prep 2: W_eff = WvSum @ WoSum, f16, WMMA-B fragment order -------
// Wp[((k>>5)*512 + n)*32 + (k&31)] = W_eff[k][n] : a 32x16 B fragment is
// 32 contiguous bytes per lane (two aligned b128 loads).
__global__ void attn_prep_weff(const float* __restrict__ WvSum,
                               const float* __restrict__ WoSum,
                               _Float16* __restrict__ Wp) {
  int h = blockIdx.x;    // K index 0..511
  int j = threadIdx.x;   // N index 0..511
  float s = 0.f;
  #pragma unroll 8
  for (int d = 0; d < 64; ++d) s += WvSum[h * 64 + d] * WoSum[d * 512 + j];
  int kb = h >> 5, kk = h & 31;
  Wp[((size_t)kb * 512 + j) * 32 + kk] = (_Float16)s;
}

// ---------- async global -> LDS (ASYNCcnt path, ISA ch.10/§08) --------------
__device__ __forceinline__ void async_b128_to_lds(unsigned lds_byte, const float* g) {
  asm volatile("global_load_async_to_lds_b128 %0, %1, off"
               :: "v"(lds_byte), "v"(g)
               : "memory");
}
__device__ __forceinline__ void wait_async0() {
  asm volatile("s_wait_asynccnt 0" ::: "memory");
}

// ---------- main GEMM: C[M,512] = A[M,512] @ Wp + bias ----------------------
__global__ __launch_bounds__(256) void attn_gemm_wmma(
    const float* __restrict__ A, const _Float16* __restrict__ Wp,
    const float* __restrict__ bias, float* __restrict__ C) {
  __shared__ float asmem[2][BUFELT];           // ping-pong 64x32 f32 tiles

  const int tid  = threadIdx.x;
  const int wave = tid >> 5;
  const int lane = tid & 31;
  const int l15  = lane & 15;
  const int hi   = lane >> 4;                  // lane half (0/1)
  const int bn   = (blockIdx.x & 1) * BN;
  const long bm  = (long)(blockIdx.x >> 1) * BM;
  const int wm   = (wave >> 2) * 32;           // wave row offset in block tile
  const int wn   = (wave & 3) * 64;            // wave col offset in block tile

  // Per-thread staging assignment: rows r0 and r0+32, one float4 column group.
  const int r0 = tid >> 3;
  const int cc = (tid & 7) << 2;
  const float* gptr0 = A + (bm + r0) * KDIM + cc;
  const float* gptr1 = A + (bm + r0 + 32) * KDIM + cc;
  unsigned lofs[2][2];
  #pragma unroll
  for (int b = 0; b < 2; ++b) {
    lofs[b][0] = (unsigned)(uintptr_t)&asmem[b][r0 * LDSW + cc];
    lofs[b][1] = (unsigned)(uintptr_t)&asmem[b][(r0 + 32) * LDSW + cc];
  }

  const v8f vzero = {0.f, 0.f, 0.f, 0.f, 0.f, 0.f, 0.f, 0.f};
  v8f acc[2][4];
  #pragma unroll
  for (int mi = 0; mi < 2; ++mi)
    #pragma unroll
    for (int ni = 0; ni < 4; ++ni) acc[mi][ni] = vzero;

  // One K-step of compute from buffer `buf` (compile-time), K offset k0.
  auto compute = [&](const float* __restrict__ buf, int k0) {
    const int kb = k0 >> 5;
    union { v16h v; uint4 q[2]; } bf[4];
    #pragma unroll
    for (int ni = 0; ni < 4; ++ni) {
      const _Float16* p =
          Wp + ((size_t)kb * 512 + (bn + wn + ni * 16 + l15)) * 32 + hi * 16;
      bf[ni].q[0] = *(const uint4*)(p);
      bf[ni].q[1] = *(const uint4*)(p + 8);
    }
    // A fragments, f32 -> f16, contiguous k order (matches Wp B packing):
    // lanes 0-15: K 0-15 ; lanes 16-31: K 16-31
    union { v16h v; _Float16 h[16]; } af[2];
    #pragma unroll
    for (int mi = 0; mi < 2; ++mi) {
      const float* row = &buf[(wm + mi * 16 + l15) * LDSW + hi * 16];
      #pragma unroll
      for (int i = 0; i < 8; ++i) {
        af[mi].h[i]     = (_Float16)row[i];
        af[mi].h[8 + i] = (_Float16)row[8 + i];
      }
    }
    #pragma unroll
    for (int mi = 0; mi < 2; ++mi)
      #pragma unroll
      for (int ni = 0; ni < 4; ++ni)
        acc[mi][ni] = __builtin_amdgcn_wmma_f32_16x16x32_f16(
            false, af[mi].v, false, bf[ni].v, (short)0, acc[mi][ni],
            false, false);
  };

  // Prologue: stage first tile into buffer 0.
  async_b128_to_lds(lofs[0][0], gptr0);
  async_b128_to_lds(lofs[0][1], gptr1);

  #pragma unroll 1
  for (int k0 = 0; k0 < KDIM; k0 += 2 * KSTEP) {
    // ---- half-step A: consume buf0, prefetch buf1 (always in range) ----
    wait_async0();
    __syncthreads();
    async_b128_to_lds(lofs[1][0], gptr0 + (k0 + KSTEP));
    async_b128_to_lds(lofs[1][1], gptr1 + (k0 + KSTEP));
    compute(asmem[0], k0);

    // ---- half-step B: consume buf1, prefetch buf0 (skip on last tile) ----
    wait_async0();
    __syncthreads();
    if (k0 + 2 * KSTEP < KDIM) {
      async_b128_to_lds(lofs[0][0], gptr0 + (k0 + 2 * KSTEP));
      async_b128_to_lds(lofs[0][1], gptr1 + (k0 + 2 * KSTEP));
    }
    compute(asmem[1], k0 + KSTEP);
  }

  // ---- epilogue: C/D layout => VGPR r holds row r (lanes 0-15) / r+8 ----
  #pragma unroll
  for (int mi = 0; mi < 2; ++mi) {
    const long rowbase = bm + wm + mi * 16 + hi * 8;
    #pragma unroll
    for (int ni = 0; ni < 4; ++ni) {
      const int col = bn + wn + ni * 16 + l15;
      const float bb = bias[col];
      #pragma unroll
      for (int r = 0; r < 8; ++r)
        C[(rowbase + r) * NDIM + col] = acc[mi][ni][r] + bb;
    }
  }
}

extern "C" void kernel_launch(void* const* d_in, const int* in_sizes, int n_in,
                              void* d_out, int out_size, void* d_ws, size_t ws_size,
                              hipStream_t stream) {
  (void)in_sizes; (void)n_in; (void)out_size; (void)ws_size;
  const float* tokens = (const float*)d_in[0];
  const float* Wv     = (const float*)d_in[3];
  const float* Wo     = (const float*)d_in[4];
  const float* bo     = (const float*)d_in[5];
  float* out = (float*)d_out;

  // workspace: WvSum (128KB) | WoSum (128KB) | Wp f16 (512KB)
  float*    wvsum = (float*)d_ws;
  float*    wosum = wvsum + 512 * 64;
  _Float16* wp    = (_Float16*)(wosum + 64 * 512);

  attn_prep_sums<<<256, 256, 0, stream>>>(Wv, Wo, wvsum, wosum);
  attn_prep_weff<<<512, 512, 0, stream>>>(wvsum, wosum, wp);
  attn_gemm_wmma<<<(MTOT / BM) * (NDIM / BN), 256, 0, stream>>>(tokens, wp, bo, out);
}
